// VQVAE_12034498363494
// MI455X (gfx1250) — matrix operand order repaired and checked by reading, not compiled, not run on hardware
//
#include <hip/hip_runtime.h>
#include <hip/hip_bf16.h>
#include <math.h>

// ---------------------------------------------------------------------------
// VQ-VAE forward for MI455X (gfx1250): f16 WMMA, LDS double-buffered GEMMs.
// ---------------------------------------------------------------------------

typedef _Float16 h16;
typedef __attribute__((ext_vector_type(8)))  _Float16 v8h;
typedef __attribute__((ext_vector_type(16))) _Float16 v16h;
typedef __attribute__((ext_vector_type(8)))  float    v8f;

#define BATCH      65536
#define SDIM       64
#define ADIM       32
#define LDIM       64
#define HDIM       1024
#define NEMB       1024

// ---------------------------------------------------------------------------
// Block-tiled WMMA GEMM:  C[M,N] = act(A[M,K] @ B[K,N] + bias)
//   A  : row-major f16 [M,K]
//   Bt : B pre-transposed to [N,K] f16 (so k-runs are contiguous)
// Tile: BM=128, BN=128, BK=32. Block = 256 threads = 8 waves (4 m x 2 n);
// each wave owns a 32x64 accumulator patch = 8 WMMA 16x16 tiles.
// Ping-pong double-buffered LDS staging: compute reads buffer p while the
// register-prefetched next k-slice is published to buffer p^1 -> exactly one
// barrier per k-step. Rows pitch-padded (40 halves) for LDS bank spread.
// Requires: M % 128 == 0, K % 32 == 0. N arbitrary (clamped loads, guarded
// stores). All barriers / WMMAs execute under block-uniform control flow.
// ACT: 0 = none, 1 = relu, 2 = tanh (MAX_ACTION == 1.0)
// ---------------------------------------------------------------------------
template <int ACT>
__global__ __launch_bounds__(256) void gemm_f16_wmma_tiled(
    const h16* __restrict__ A, const h16* __restrict__ Bt,
    const float* __restrict__ bias,
    float* __restrict__ outF32, h16* __restrict__ outF16,
    int M, int N, int K)
{
    constexpr int BM = 128, BN = 128, BK = 32;
    constexpr int PITCH = BK + 8;                      // 40 halves = 80 B
    __shared__ __align__(16) h16 sA[2][BM * PITCH];    // 2 x 10 KB
    __shared__ __align__(16) h16 sB[2][BN * PITCH];    // 2 x 10 KB

    const int tid     = threadIdx.x;
    const int lane    = tid & 31;
    const int wave    = tid >> 5;
    const int wm      = wave >> 1;                  // 0..3  (m direction)
    const int wn      = wave & 1;                   // 0..1  (n direction)
    const int m0      = blockIdx.x * BM;
    const int n0      = blockIdx.y * BN;
    const int half_id = lane >> 4;
    const int r       = lane & 15;

    // ---- staging assignment: each thread moves 16 halves of A and B per slice
    const int sr = tid >> 1;                        // 0..127 : A row / B col
    const int sc = (tid & 1) * 16;                  // 0 or 16 within k-slice
    int gb_col = n0 + sr; if (gb_col > N - 1) gb_col = N - 1;
    const h16* gA = A  + (size_t)(m0 + sr) * K + sc;
    const h16* gB = Bt + (size_t)gb_col   * K + sc;
    const int soff = sr * PITCH + sc;

    v8f acc[2][4] = {};                             // [m-tile][n-tile]

    const int nsteps = K / BK;

    // stage slice 0 into buffer 0
    v16h ra = *(const v16h*)(gA);
    v16h rb = *(const v16h*)(gB);
    *(v16h*)(&sA[0][soff]) = ra;
    *(v16h*)(&sB[0][soff]) = rb;
    __syncthreads();

    int p = 0;
    for (int s = 0; s < nsteps; ++s) {
        const bool more = (s + 1 < nsteps);         // block-uniform
        // register-prefetch next k-slice while computing on the current one
        if (more) {
            ra = *(const v16h*)(gA + (s + 1) * BK);
            rb = *(const v16h*)(gB + (s + 1) * BK);
        }
        if (s + 2 < nsteps) {
            __builtin_prefetch(gA + (s + 2) * BK, 0, 0);
            __builtin_prefetch(gB + (s + 2) * BK, 0, 0);
        }

        const h16* cA = sA[p];
        const h16* cB = sB[p];

        // B fragments (shared across both m-tiles of this wave)
        v16h bf[4];
#pragma unroll
        for (int nt = 0; nt < 4; ++nt) {
            int col = wn * 64 + nt * 16 + r;
            union { v16h v; v8h h[2]; } bu;
            bu.h[0] = *(const v8h*)(cB + col * PITCH + half_id * 8);
            bu.h[1] = *(const v8h*)(cB + col * PITCH + 16 + half_id * 8);
            bf[nt] = bu.v;
        }
#pragma unroll
        for (int mt = 0; mt < 2; ++mt) {
            int row = wm * 32 + mt * 16 + r;
            union { v16h v; v8h h[2]; } af;
            af.h[0] = *(const v8h*)(cA + row * PITCH + half_id * 8);
            af.h[1] = *(const v8h*)(cA + row * PITCH + 16 + half_id * 8);
#pragma unroll
            for (int nt = 0; nt < 4; ++nt)
                acc[mt][nt] = __builtin_amdgcn_wmma_f32_16x16x32_f16(
                    false, af.v, false, bf[nt], (short)0, acc[mt][nt], false, false);
        }

        // publish next slice into the other buffer; one barrier per step
        if (more) {
            *(v16h*)(&sA[p ^ 1][soff]) = ra;
            *(v16h*)(&sB[p ^ 1][soff]) = rb;
            __syncthreads();
        }
        p ^= 1;
    }

    // ---- epilogue: C layout = 8 VGPRs; lane<16 -> M=v, lane>=16 -> M=8+v
#pragma unroll
    for (int mt = 0; mt < 2; ++mt) {
#pragma unroll
        for (int nt = 0; nt < 4; ++nt) {
            int n = n0 + wn * 64 + nt * 16 + r;
            if (n < N) {
                float bv = bias ? bias[n] : 0.0f;
#pragma unroll
                for (int v = 0; v < 8; ++v) {
                    int   m = m0 + wm * 32 + mt * 16 + v + half_id * 8;
                    float x = acc[mt][nt][v] + bv;
                    if (ACT == 1) x = x > 0.0f ? x : 0.0f;
                    else if (ACT == 2) x = tanhf(x);
                    size_t o = (size_t)m * N + n;
                    if (outF32) outF32[o] = x;
                    if (outF16) outF16[o] = (h16)x;
                }
            }
        }
    }
}

// ---------------------------------------------------------------------------
// VQ: scores = ||c||^2 - 2 * (z @ codebook^T)  via WMMA, then per-row argmin.
// Block = 512 threads = 16 waves; covers a 16x1024 score tile in LDS (64 KB),
// then wave w reduces row w with a strided scan + shfl reduction (wave32).
// ---------------------------------------------------------------------------
__global__ __launch_bounds__(512) void vq_argmin(
    const h16* __restrict__ Z, const h16* __restrict__ CB,
    const float* __restrict__ cnorm, int* __restrict__ idxOut)
{
    __shared__ float sc[16][NEMB];             // 64 KB
    const int lane    = threadIdx.x & 31;
    const int wave    = threadIdx.x >> 5;      // 0..15
    const int m0      = blockIdx.x * 16;
    const int n0      = wave * 64;
    const int half_id = lane >> 4;
    const int r       = lane & 15;

    v8f acc[4] = {};
    const h16* arow = Z + (size_t)(m0 + r) * LDIM;
#pragma unroll
    for (int k0 = 0; k0 < LDIM; k0 += 32) {
        union { v16h v; v8h h[2]; } af;
        af.h[0] = *(const v8h*)(arow + k0 + half_id * 8);
        af.h[1] = *(const v8h*)(arow + k0 + 16 + half_id * 8);
#pragma unroll
        for (int t = 0; t < 4; ++t) {
            int  n  = n0 + t * 16 + r;
            union { v16h v; v8h h[2]; } bu;
            bu.h[0] = *(const v8h*)(CB + (size_t)n * LDIM + k0 + half_id * 8);
            bu.h[1] = *(const v8h*)(CB + (size_t)n * LDIM + k0 + 16 + half_id * 8);
            v16h bf = bu.v;
            acc[t] = __builtin_amdgcn_wmma_f32_16x16x32_f16(
                false, af.v, false, bf, (short)0, acc[t], false, false);
        }
    }
#pragma unroll
    for (int t = 0; t < 4; ++t) {
        int   n  = n0 + t * 16 + r;
        float cn = cnorm[n];
#pragma unroll
        for (int v = 0; v < 8; ++v)
            sc[v + half_id * 8][n] = cn - 2.0f * acc[t][v];
    }
    __syncthreads();

    // wave `wave` reduces row `wave`; keep first index on ties (argmin semantics)
    float best = 3.0e38f;
    int   bi   = 0;
    const float* rowp = sc[wave];
    for (int c = lane; c < NEMB; c += 32) {
        float v = rowp[c];
        if (v < best) { best = v; bi = c; }
    }
#pragma unroll
    for (int off = 16; off > 0; off >>= 1) {
        float ob = __shfl_down(best, off, 32);
        int   oi = __shfl_down(bi,   off, 32);
        if (ob < best || (ob == best && oi < bi)) { best = ob; bi = oi; }
    }
    if (lane == 0) idxOut[m0 + wave] = bi;
}

// ---------------------------------------------------------------------------
// Gather zq, mean/logstd GEMVs (f32), reparameterize, pack decoder input.
// Block = 256 threads = 4 rows x 64 latent dims.
// ---------------------------------------------------------------------------
__global__ __launch_bounds__(256) void vq_reparam(
    const int* __restrict__ idx, const float* __restrict__ cb,
    const float* __restrict__ mean_w, const float* __restrict__ mean_b,
    const float* __restrict__ logstd_w, const float* __restrict__ logstd_b,
    const float* __restrict__ eps, const float* __restrict__ state,
    float* __restrict__ outMean, float* __restrict__ outStd,
    h16* __restrict__ xcat)
{
    __shared__ float zq[4][LDIM];
    const int rr = threadIdx.x >> 6;           // row within block
    const int j  = threadIdx.x & 63;
    const int b  = blockIdx.x * 4 + rr;

    zq[rr][j] = cb[(size_t)idx[b] * LDIM + j];
    __syncthreads();

    float m  = mean_b[j];
    float ls = logstd_b[j];
#pragma unroll 8
    for (int k = 0; k < LDIM; ++k) {
        float z = zq[rr][k];
        m  += z * mean_w[k * LDIM + j];
        ls += z * logstd_w[k * LDIM + j];
    }
    ls = fminf(fmaxf(ls, -4.0f), 15.0f);
    float sd = expf(ls);
    float zz = m + sd * eps[(size_t)b * LDIM + j];

    outMean[(size_t)b * LDIM + j] = m;
    outStd [(size_t)b * LDIM + j] = sd;
    xcat[(size_t)b * 128 + j]      = (h16)state[(size_t)b * SDIM + j];
    xcat[(size_t)b * 128 + 64 + j] = (h16)zz;
}

// ---------------------------------------------------------------------------
// Prep kernels
// ---------------------------------------------------------------------------
__global__ void pack_enc_input(const float* __restrict__ state,
                               const float* __restrict__ action,
                               h16* __restrict__ incat, int total)
{
    int t = blockIdx.x * blockDim.x + threadIdx.x;
    if (t >= total) return;
    int b = t / 96, j = t - b * 96;
    float v = (j < 64) ? state[(size_t)b * SDIM + j]
                       : action[(size_t)b * ADIM + (j - 64)];
    incat[t] = (h16)v;
}

__global__ void transpose_to_f16(const float* __restrict__ w, h16* __restrict__ wt,
                                 int K, int N, int total)
{
    int t = blockIdx.x * blockDim.x + threadIdx.x;
    if (t >= total) return;
    int k = t / N, n = t - k * N;
    wt[(size_t)n * K + k] = (h16)w[t];         // [K,N] row-major -> [N,K]
}

__global__ void codebook_prep(const float* __restrict__ cb,
                              h16* __restrict__ cb16, float* __restrict__ cnorm)
{
    int j = blockIdx.x * blockDim.x + threadIdx.x;   // 0..1023
    if (j >= NEMB) return;
    float s = 0.0f;
    for (int k = 0; k < LDIM; ++k) {
        float v = cb[(size_t)j * LDIM + k];
        cb16[(size_t)j * LDIM + k] = (h16)v;
        s += v * v;
    }
    cnorm[j] = s;
}

// ---------------------------------------------------------------------------
// Host-side launcher
// ---------------------------------------------------------------------------
extern "C" void kernel_launch(void* const* d_in, const int* in_sizes, int n_in,
                              void* d_out, int out_size, void* d_ws, size_t ws_size,
                              hipStream_t stream)
{
    (void)in_sizes; (void)n_in; (void)out_size; (void)ws_size;

    const float* state    = (const float*)d_in[0];
    const float* action   = (const float*)d_in[1];
    const float* eps      = (const float*)d_in[2];
    const float* enc_w1   = (const float*)d_in[3];
    const float* enc_b1   = (const float*)d_in[4];
    const float* enc_w2   = (const float*)d_in[5];
    const float* enc_b2   = (const float*)d_in[6];
    const float* enc_w3   = (const float*)d_in[7];
    const float* enc_b3   = (const float*)d_in[8];
    const float* mean_w   = (const float*)d_in[9];
    const float* mean_b   = (const float*)d_in[10];
    const float* logstd_w = (const float*)d_in[11];
    const float* logstd_b = (const float*)d_in[12];
    const float* codebook = (const float*)d_in[13];
    const float* dec_w1   = (const float*)d_in[14];
    const float* dec_b1   = (const float*)d_in[15];
    const float* dec_w2   = (const float*)d_in[16];
    const float* dec_b2   = (const float*)d_in[17];
    const float* dec_w3   = (const float*)d_in[18];
    const float* dec_b3   = (const float*)d_in[19];

    float* out_u    = (float*)d_out;                       // [B,32]
    float* out_mean = out_u    + (size_t)BATCH * ADIM;     // [B,64]
    float* out_std  = out_mean + (size_t)BATCH * LDIM;     // [B,64]

    // ---- carve workspace ----
    char*  ws  = (char*)d_ws;
    size_t off = 0;
    auto carve = [&](size_t bytes) -> char* {
        char* p = ws + off;
        off += (bytes + 255) & ~(size_t)255;
        return p;
    };
    h16*   wT_e1   = (h16*)carve((size_t)96  * HDIM * 2);
    h16*   wT_e2   = (h16*)carve((size_t)HDIM * HDIM * 2);
    h16*   wT_e3   = (h16*)carve((size_t)HDIM * LDIM * 2);
    h16*   wT_d1   = (h16*)carve((size_t)128 * HDIM * 2);
    h16*   wT_d2   = (h16*)carve((size_t)HDIM * HDIM * 2);
    h16*   wT_d3   = (h16*)carve((size_t)HDIM * ADIM * 2);
    h16*   cb16    = (h16*)carve((size_t)NEMB * LDIM * 2);
    float* cnorm   = (float*)carve((size_t)NEMB * 4);
    h16*   incat   = (h16*)carve((size_t)BATCH * 96  * 2);
    h16*   xcat    = (h16*)carve((size_t)BATCH * 128 * 2);
    h16*   midz    = (h16*)carve((size_t)BATCH * LDIM * 2);
    int*   idx     = (int*)carve((size_t)BATCH * 4);
    h16*   bufA    = (h16*)carve((size_t)BATCH * HDIM * 2);
    h16*   bufB    = (h16*)carve((size_t)BATCH * HDIM * 2);

    // ---- prep: weight transpose+convert, codebook, input pack ----
    auto tgrid = [](int total) { return dim3((total + 255) / 256); };
    transpose_to_f16<<<tgrid(96 * HDIM),   256, 0, stream>>>(enc_w1, wT_e1, 96,   HDIM, 96 * HDIM);
    transpose_to_f16<<<tgrid(HDIM * HDIM), 256, 0, stream>>>(enc_w2, wT_e2, HDIM, HDIM, HDIM * HDIM);
    transpose_to_f16<<<tgrid(HDIM * LDIM), 256, 0, stream>>>(enc_w3, wT_e3, HDIM, LDIM, HDIM * LDIM);
    transpose_to_f16<<<tgrid(128 * HDIM),  256, 0, stream>>>(dec_w1, wT_d1, 128,  HDIM, 128 * HDIM);
    transpose_to_f16<<<tgrid(HDIM * HDIM), 256, 0, stream>>>(dec_w2, wT_d2, HDIM, HDIM, HDIM * HDIM);
    transpose_to_f16<<<tgrid(HDIM * ADIM), 256, 0, stream>>>(dec_w3, wT_d3, HDIM, ADIM, HDIM * ADIM);
    codebook_prep<<<dim3(NEMB / 256), 256, 0, stream>>>(codebook, cb16, cnorm);
    pack_enc_input<<<tgrid(BATCH * 96), 256, 0, stream>>>(state, action, incat, BATCH * 96);

    const int MB = BATCH / 128;                 // 512 row blocks
    auto ggrid = [&](int N) { return dim3(MB, (N + 127) / 128); };

    // ---- encoder ----
    gemm_f16_wmma_tiled<1><<<ggrid(HDIM), 256, 0, stream>>>(incat, wT_e1, enc_b1, nullptr, bufA, BATCH, HDIM, 96);
    gemm_f16_wmma_tiled<1><<<ggrid(HDIM), 256, 0, stream>>>(bufA,  wT_e2, enc_b2, nullptr, bufB, BATCH, HDIM, HDIM);
    gemm_f16_wmma_tiled<1><<<ggrid(LDIM), 256, 0, stream>>>(bufB,  wT_e3, enc_b3, nullptr, midz, BATCH, LDIM, HDIM);

    // ---- VQ argmin + reparameterization ----
    vq_argmin<<<dim3(BATCH / 16), 512, 0, stream>>>(midz, cb16, cnorm, idx);
    vq_reparam<<<dim3(BATCH / 4), 256, 0, stream>>>(idx, codebook, mean_w, mean_b,
                                                    logstd_w, logstd_b, eps, state,
                                                    out_mean, out_std, xcat);

    // ---- decoder ----
    gemm_f16_wmma_tiled<1><<<ggrid(HDIM), 256, 0, stream>>>(xcat, wT_d1, dec_b1, nullptr, bufA, BATCH, HDIM, 128);
    gemm_f16_wmma_tiled<1><<<ggrid(HDIM), 256, 0, stream>>>(bufA, wT_d2, dec_b2, nullptr, bufB, BATCH, HDIM, HDIM);
    gemm_f16_wmma_tiled<2><<<ggrid(ADIM), 256, 0, stream>>>(bufB, wT_d3, dec_b3, out_u, nullptr, BATCH, ADIM, HDIM);
}
